// GRUS_2_26843545600091
// MI455X (gfx1250) — matrix operand order repaired and checked by reading, not compiled, not run on hardware
//
#include <hip/hip_runtime.h>

typedef __attribute__((ext_vector_type(16))) __bf16 v16bf;
typedef __attribute__((ext_vector_type(8)))  float  v8f;
typedef __attribute__((ext_vector_type(4)))  float  v4f;
typedef __attribute__((ext_vector_type(4)))  unsigned int v4u;

constexpr int kD    = 256;
constexpr int kN    = 1024;
constexpr int kP    = 32;
constexpr int kL    = 3;
constexpr int kR    = 32;
constexpr int kM    = kN * kP;   // 32768 rows (n*P+p)
constexpr int kNRel = 1000;

__device__ __forceinline__ unsigned short f2bf(float f) {
  unsigned int u = __float_as_uint(f);
  u += 0x7FFFu + ((u >> 16) & 1u);   // round-to-nearest-even
  return (unsigned short)(u >> 16);
}

// ---------------- zero fill (h := 0 before step 0) ----------------
__global__ void k_zero(float* __restrict__ p, int n) {
  int i = blockIdx.x * blockDim.x + threadIdx.x;
  if (i < n) p[i] = 0.0f;
}

// ---------------- fp32 -> bf16 weight convert ----------------
__global__ void k_cvt(const float* __restrict__ a, unsigned short* __restrict__ b, int n) {
  int i = blockIdx.x * blockDim.x + threadIdx.x;
  if (i < n) b[i] = f2bf(a[i]);
}

// ---------------- PE[r,e] = sum_d rel_emb[r,d]*we_w[e,d]; PD likewise ----------------
__global__ void k_proj(const float* __restrict__ rel_emb,
                       const float* __restrict__ we_w,
                       const float* __restrict__ wd_w,
                       float* __restrict__ PE, float* __restrict__ PD) {
  __shared__ float srel[kD];
  const int r = blockIdx.x;          // 0..999
  const int t = threadIdx.x;         // 0..255 -> output element e
  srel[t] = rel_emb[r * kD + t];
  __syncthreads();
  const float* we = we_w + (size_t)t * kD;
  const float* wd = wd_w + (size_t)t * kD;
  float pe = 0.0f, pd = 0.0f;
#pragma unroll 8
  for (int d = 0; d < kD; ++d) {
    float rv = srel[d];
    pe += rv * we[d];
    pd += rv * wd[d];
  }
  PE[(size_t)r * kD + t] = pe;
  PD[(size_t)r * kD + t] = pd;
}

// ---------------- att[n,r] = softmax_r( rel_emb[srs[n,r]] . wu_w + wu_b ) ----------------
__global__ void k_att(const float* __restrict__ rel_emb,
                      const float* __restrict__ wu_w,
                      const float* __restrict__ wu_b,
                      const int* __restrict__ srs,
                      float* __restrict__ att) {
  const int wave = threadIdx.x >> 5, lane = threadIdx.x & 31;
  const int n = blockIdx.x * 8 + wave;          // grid = kN/8
  const int rel = srs[n * kR + lane];
  const float* rp = rel_emb + (size_t)rel * kD;
  float s = wu_b[0];
#pragma unroll 8
  for (int d = 0; d < kD; ++d) s += rp[d] * wu_w[d];
  float m = s;
  for (int o = 16; o > 0; o >>= 1) m = fmaxf(m, __shfl_xor(m, o, 32));
  float e = expf(s - m);
  float sum = e;
  for (int o = 16; o > 0; o >>= 1) sum += __shfl_xor(sum, o, 32);
  att[n * kR + lane] = e / sum;
}

// ---------------- build X[l][m][d] (bf16): x = r_emb - 0.001*lrelu(dl*PD + a*PE + wd_b + we_b)
__global__ void k_buildx(const float* __restrict__ ent_emb,
                         const float* __restrict__ rel_emb,
                         const float* __restrict__ wd_b,
                         const float* __restrict__ we_b,
                         const float* __restrict__ PE,
                         const float* __restrict__ PD,
                         const float* __restrict__ att,
                         const int* __restrict__ pair,
                         const int* __restrict__ spi,
                         const int* __restrict__ spe,
                         const int* __restrict__ srs,
                         unsigned short* __restrict__ X) {
  const int wave = threadIdx.x >> 5, lane = threadIdx.x & 31;
  const int m = blockIdx.x * 8 + wave;         // grid = kM/8
  const int n = m >> 5, p = m & 31;
  const int head = pair[n * 2 + 0], tail = pair[n * 2 + 1];
  const int e0i = spe[(size_t)m * 2 + 0], e1i = spe[(size_t)m * 2 + 1];

  float hh = 0, tt = 0, ht = 0, h0 = 0, t0 = 0, a00 = 0, h1 = 0, t1 = 0, b11 = 0;
#pragma unroll
  for (int i = 0; i < 8; ++i) {
    const int e = i * 32 + lane;
    const float H = ent_emb[(size_t)head * kD + e];
    const float T = ent_emb[(size_t)tail * kD + e];
    const float A = ent_emb[(size_t)e0i * kD + e];
    const float B = ent_emb[(size_t)e1i * kD + e];
    hh += H * H; tt += T * T; ht += H * T;
    h0 += H * A; t0 += T * A; a00 += A * A;
    h1 += H * B; t1 += T * B; b11 += B * B;
  }
  for (int o = 16; o > 0; o >>= 1) {
    hh += __shfl_xor(hh, o, 32); tt += __shfl_xor(tt, o, 32); ht += __shfl_xor(ht, o, 32);
    h0 += __shfl_xor(h0, o, 32); t0 += __shfl_xor(t0, o, 32); a00 += __shfl_xor(a00, o, 32);
    h1 += __shfl_xor(h1, o, 32); t1 += __shfl_xor(t1, o, 32); b11 += __shfl_xor(b11, o, 32);
  }
  const float eps = 1e-8f;
  const float nh = fmaxf(sqrtf(hh), eps), nt = fmaxf(sqrtf(tt), eps);
  const float n0 = fmaxf(sqrtf(a00), eps), n1 = fmaxf(sqrtf(b11), eps);
  const float sht  = ht / (nh * nt);
  const float sim1 = 0.5f * (h0 / (nh * n0) + t0 / (nt * n0));
  const float sim2 = 0.5f * (h1 / (nh * n1) + t1 / (nt * n1));
  float dv[3];
  dv[0] = 1.0f - 0.5f * (sht + sim1);
  dv[1] = 1.0f - 0.5f * (sim1 + sim2);
  dv[2] = 1.0f - 0.5f * (sim2 + sht);

  for (int l = 0; l < kL; ++l) {
    const int fi  = spi[(size_t)m * kL + l];
    const int rel = srs[n * kR + fi];
    const float a  = att[n * kR + fi];
    const float dl = dv[l];
    const float* rp = rel_emb + (size_t)rel * kD;
    const float* pd = PD + (size_t)rel * kD;
    const float* pe = PE + (size_t)rel * kD;
    unsigned short* xo = X + ((size_t)l * kM + m) * kD;
#pragma unroll
    for (int i = 0; i < 8; ++i) {
      const int e = i * 32 + lane;
      const float v = dl * pd[e] + a * pe[e] + wd_b[e] + we_b[e];
      const float lr = v > 0.0f ? v : 0.01f * v;
      xo[e] = f2bf(rp[e] - 0.001f * lr);
    }
  }
}

// ---------------- one GRU step: h = GRUCell(X_k, h) using bf16 WMMA ----------------
// Block: 256 threads = 8 waves, 128 rows of h. 16 column tiles of 16.
// LDS: 64KB h snapshot + 48KB for all six weight tiles (r,z,n x ih,hh) = 112KB.
__global__ __launch_bounds__(256) void k_gru(const unsigned short* __restrict__ Xk,
                                             const unsigned short* __restrict__ Wih,
                                             const unsigned short* __restrict__ Whh,
                                             const float* __restrict__ b_ih,
                                             const float* __restrict__ b_hh,
                                             float* __restrict__ h) {
  __shared__ __align__(16) unsigned short sH[128][kD];     // 64 KB: bf16 snapshot of h rows
  __shared__ __align__(16) unsigned short sW[6][16][kD];   // 48 KB: 6 weight tiles

  const int tid   = threadIdx.x;
  const int wave  = tid >> 5;
  const int lane  = tid & 31;
  const int rsel  = lane & 15;      // A: row-in-tile, B: col-in-tile, C/D: col
  const int hi    = lane >> 4;      // lane half
  const int mBase = blockIdx.x * 128;
  const int mW    = mBase + wave * 16;

  // snapshot h (fp32 -> bf16) into LDS
  {
    const float* hb = h + (size_t)mBase * kD;
    unsigned short* sHf = &sH[0][0];
    for (int idx = tid; idx < 128 * kD / 4; idx += 256) {
      v4f q = *(const v4f*)(hb + (size_t)idx * 4);
      const int off = idx * 4;
      sHf[off + 0] = f2bf(q.x);
      sHf[off + 1] = f2bf(q.y);
      sHf[off + 2] = f2bf(q.z);
      sHf[off + 3] = f2bf(q.w);
    }
  }
  __syncthreads();

  for (int ct = 0; ct < 16; ++ct) {
    const int c = ct * 16;
    v8f acc0 = {0,0,0,0,0,0,0,0}, acc1 = {0,0,0,0,0,0,0,0}, acc2 = {0,0,0,0,0,0,0,0};
    v8f acc3 = {0,0,0,0,0,0,0,0}, acc4 = {0,0,0,0,0,0,0,0}, acc5 = {0,0,0,0,0,0,0,0};

    // ---- stage all six 16-col weight tiles for this column tile ----
    __syncthreads();   // protect sW against consumers from previous iteration
    for (int id = tid; id < 6 * 16 * 32; id += 256) {
      const int t  = id >> 9;           // 0..5 (0..2: w_ih r/z/n, 3..5: w_hh r/z/n)
      const int rm = id & 511;
      const int rr = rm >> 5;           // row within tile
      const int ch = rm & 31;           // 8-element chunk
      const unsigned short* base = (t < 3) ? Wih : Whh;
      const int g = (t < 3) ? t : (t - 3);
      *(v4u*)&sW[t][rr][ch * 8] =
          *(const v4u*)(base + ((size_t)(g * 256 + c + rr)) * kD + ch * 8);
    }
    __syncthreads();

    // ---- K loop: 2 A-loads + 6 B-loads + 6 WMMAs per iteration ----
#pragma unroll
    for (int k = 0; k < 8; ++k) {
      const int eA = k * 32 + hi * 16;   // A: one 16-elt K chunk
      const int eB = k * 32 + hi * 16;   // B: one 16-elt K chunk
      union { v16bf v; v4u q[2]; } ua, uh, ub;
      const unsigned short* xp = Xk + (size_t)(mW + rsel) * kD + eA;
      ua.q[0] = *(const v4u*)xp;
      ua.q[1] = *(const v4u*)(xp + 8);
      uh.q[0] = *(const v4u*)&sH[wave * 16 + rsel][eA];
      uh.q[1] = *(const v4u*)&sH[wave * 16 + rsel][eA + 8];

      ub.q[0] = *(const v4u*)&sW[0][rsel][eB];
      ub.q[1] = *(const v4u*)&sW[0][rsel][eB + 8];
      acc0 = __builtin_amdgcn_wmma_f32_16x16x32_bf16(false, ua.v, false, ub.v, (short)0, acc0, false, false);
      ub.q[0] = *(const v4u*)&sW[1][rsel][eB];
      ub.q[1] = *(const v4u*)&sW[1][rsel][eB + 8];
      acc1 = __builtin_amdgcn_wmma_f32_16x16x32_bf16(false, ua.v, false, ub.v, (short)0, acc1, false, false);
      ub.q[0] = *(const v4u*)&sW[2][rsel][eB];
      ub.q[1] = *(const v4u*)&sW[2][rsel][eB + 8];
      acc2 = __builtin_amdgcn_wmma_f32_16x16x32_bf16(false, ua.v, false, ub.v, (short)0, acc2, false, false);

      ub.q[0] = *(const v4u*)&sW[3][rsel][eB];
      ub.q[1] = *(const v4u*)&sW[3][rsel][eB + 8];
      acc3 = __builtin_amdgcn_wmma_f32_16x16x32_bf16(false, uh.v, false, ub.v, (short)0, acc3, false, false);
      ub.q[0] = *(const v4u*)&sW[4][rsel][eB];
      ub.q[1] = *(const v4u*)&sW[4][rsel][eB + 8];
      acc4 = __builtin_amdgcn_wmma_f32_16x16x32_bf16(false, uh.v, false, ub.v, (short)0, acc4, false, false);
      ub.q[0] = *(const v4u*)&sW[5][rsel][eB];
      ub.q[1] = *(const v4u*)&sW[5][rsel][eB + 8];
      acc5 = __builtin_amdgcn_wmma_f32_16x16x32_bf16(false, uh.v, false, ub.v, (short)0, acc5, false, false);
    }

    // ---- gates + in-place h update for this column tile ----
    const int colg = c + rsel;
    const float bir = b_ih[colg],       bhr = b_hh[colg];
    const float biz = b_ih[256 + colg], bhz = b_hh[256 + colg];
    const float bin = b_ih[512 + colg], bhn = b_hh[512 + colg];
#pragma unroll
    for (int j = 0; j < 8; ++j) {
      const int m = mW + j + (hi << 3);     // C/D layout: row = j + 8*hi, col = lane&15
      const float xr = acc0[j] + bir, xz = acc1[j] + biz, xn = acc2[j] + bin;
      const float hr = acc3[j] + bhr, hz = acc4[j] + bhz, hn = acc5[j] + bhn;
      const float r  = 1.0f / (1.0f + expf(-(xr + hr)));
      const float z  = 1.0f / (1.0f + expf(-(xz + hz)));
      const float nv = tanhf(xn + r * hn);
      const float hold = h[(size_t)m * kD + colg];
      h[(size_t)m * kD + colg] = (1.0f - z) * nv + z * hold;
    }
  }
}

extern "C" void kernel_launch(void* const* d_in, const int* in_sizes, int n_in,
                              void* d_out, int out_size, void* d_ws, size_t ws_size,
                              hipStream_t stream) {
  (void)in_sizes; (void)n_in; (void)out_size; (void)ws_size;
  const float* ent_emb = (const float*)d_in[0];
  const float* rel_emb = (const float*)d_in[1];
  const float* wd_w    = (const float*)d_in[2];
  const float* wd_b    = (const float*)d_in[3];
  const float* we_w    = (const float*)d_in[4];
  const float* we_b    = (const float*)d_in[5];
  const float* wu_w    = (const float*)d_in[6];
  const float* wu_b    = (const float*)d_in[7];
  const float* w_ih    = (const float*)d_in[8];
  const float* w_hh    = (const float*)d_in[9];
  const float* b_ih    = (const float*)d_in[10];
  const float* b_hh    = (const float*)d_in[11];
  const int* pair = (const int*)d_in[12];
  const int* spi  = (const int*)d_in[13];
  const int* spe  = (const int*)d_in[14];
  const int* srs  = (const int*)d_in[15];
  float* h = (float*)d_out;

  char* ws = (char*)d_ws;
  size_t off = 0;
  unsigned short* X   = (unsigned short*)(ws + off); off += (size_t)kL * kM * kD * 2;  // 48 MB bf16
  unsigned short* Wih = (unsigned short*)(ws + off); off += (size_t)768 * kD * 2;
  unsigned short* Whh = (unsigned short*)(ws + off); off += (size_t)768 * kD * 2;
  float* PE  = (float*)(ws + off); off += (size_t)kNRel * kD * 4;
  float* PD  = (float*)(ws + off); off += (size_t)kNRel * kD * 4;
  float* att = (float*)(ws + off); off += (size_t)kN * kR * 4;

  k_zero<<<(kM * kD + 255) / 256, 256, 0, stream>>>(h, kM * kD);
  k_cvt<<<(768 * kD + 255) / 256, 256, 0, stream>>>(w_ih, Wih, 768 * kD);
  k_cvt<<<(768 * kD + 255) / 256, 256, 0, stream>>>(w_hh, Whh, 768 * kD);
  k_proj<<<kNRel, 256, 0, stream>>>(rel_emb, we_w, wd_w, PE, PD);
  k_att<<<kN / 8, 256, 0, stream>>>(rel_emb, wu_w, wu_b, srs, att);
  k_buildx<<<kM / 8, 256, 0, stream>>>(ent_emb, rel_emb, wd_b, we_b, PE, PD, att,
                                       pair, spi, spe, srs, X);
  for (int step = 0; step < kL; ++step) {
    k_gru<<<kM / 128, 256, 0, stream>>>(X + (size_t)step * kM * kD, Wih, Whh, b_ih, b_hh, h);
  }
}
